// AttentionBlock_41523743818156
// MI455X (gfx1250) — matrix operand order repaired and checked by reading, not compiled, not run on hardware
//
#include <hip/hip_runtime.h>
#include <hip/hip_bf16.h>

// Problem constants (from reference)
#define BB 2
#define SS 2048
#define DD 1024
#define HH 16
#define HD 64

typedef __attribute__((ext_vector_type(16))) __bf16 v16bf;
typedef __attribute__((ext_vector_type(8)))  float  v8f;

// Load a 16-element bf16 WMMA operand fragment as two contiguous 16B runs.
__device__ __forceinline__ v16bf load_frag(const __bf16* p0, const __bf16* p1) {
    union { v16bf v; uint4 u[2]; } t;
    t.u[0] = *reinterpret_cast<const uint4*>(p0);
    t.u[1] = *reinterpret_cast<const uint4*>(p1);
    return t.v;
}

// ---------------------------------------------------------------------------
// CDNA5 async global->LDS copy (16B per lane), tracked by ASYNCcnt.
// ---------------------------------------------------------------------------
#define USE_ASYNC_LDS 1

__device__ __forceinline__ void copy16_g2l(void* lds_ptr, const void* gptr) {
#if USE_ASYNC_LDS
    const unsigned l = (unsigned)(uintptr_t)lds_ptr;   // LDS byte offset (addr[31:0])
    asm volatile("global_load_async_to_lds_b128 %0, %1, off"
                 :: "v"(l), "v"(gptr) : "memory");
#else
    *reinterpret_cast<uint4*>(lds_ptr) = *reinterpret_cast<const uint4*>(gptr);
#endif
}
__device__ __forceinline__ void copy16_wait() {
#if USE_ASYNC_LDS
    asm volatile("s_wait_asynccnt 0" ::: "memory");
#endif
}

// ---------------------------------------------------------------------------
// One-time prep kernels (weights-stationary bf16 pipeline)
// ---------------------------------------------------------------------------
__global__ __launch_bounds__(256)
void cvt_bf16_kernel(const float* __restrict__ in, __bf16* __restrict__ out)
{
    const int i = blockIdx.x * 256 + threadIdx.x;
    const float4 v = reinterpret_cast<const float4*>(in)[i];
    union { __bf16 h[4]; uint2 u; } pk;
    pk.h[0] = (__bf16)v.x; pk.h[1] = (__bf16)v.y;
    pk.h[2] = (__bf16)v.z; pk.h[3] = (__bf16)v.w;
    reinterpret_cast<uint2*>(out)[i] = pk.u;
}

// W[K][N] f32 -> Wt[N][K] bf16, 32x32 tiles via LDS
__global__ __launch_bounds__(256)
void transpose_cvt_kernel(const float* __restrict__ W, __bf16* __restrict__ Wt,
                          int K, int N)
{
    __shared__ float tile[32][33];
    const int n0 = blockIdx.x * 32, k0 = blockIdx.y * 32;
    const int tx = threadIdx.x & 31, ty = threadIdx.x >> 5;  // 8 rows
    #pragma unroll
    for (int j = 0; j < 4; j++) {
        const int k = ty + j * 8;
        tile[k][tx] = W[(size_t)(k0 + k) * N + n0 + tx];
    }
    __syncthreads();
    #pragma unroll
    for (int j = 0; j < 4; j++) {
        const int n = ty + j * 8;
        Wt[(size_t)(n0 + n) * K + k0 + tx] = (__bf16)tile[tx][n];
    }
}

// ---------------------------------------------------------------------------
// GEMM C[M,N] = A[M,K] @ Bt[N,K]^T ; A,Bt bf16, C f32.
// Block tile 64x256, 8 waves (2x4), each wave 32x64 = 2x4 WMMA tiles.
// Double-buffered LDS; staging = pure 16B async global->LDS DMA.
// ---------------------------------------------------------------------------
#define GTM 64
#define GTN 256
#define GTK 32
#define KPAD 8   // LDS row stride GTK+8 = 40 elems = 80B (16B-aligned)

__global__ __launch_bounds__(256)
void gemm_bf16wmma(const __bf16* __restrict__ A, const __bf16* __restrict__ Bt,
                   float* __restrict__ C, int M, int N, int K)
{
    __shared__ __bf16 As [2][GTM][GTK + KPAD];   // [buf][row][k]
    __shared__ __bf16 BsT[2][GTN][GTK + KPAD];   // [buf][col][k]

    const int tid  = threadIdx.x;
    const int wave = tid >> 5;
    const int lane = tid & 31;
    const int lh   = lane & 15;
    const bool hi  = lane >= 16;
    const int wm   = wave >> 2;   // 0..1
    const int wn   = wave & 3;    // 0..3
    const int bm   = blockIdx.y * GTM;
    const int bn   = blockIdx.x * GTN;

    v8f acc[2][4] = {};

    // Per-thread staging coordinates (16B granules)
    const int ar = tid >> 2, aq = (tid & 3) * 8;          // A: 256 granules

    auto stage = [&](int k0, int buf) {
        // A tile: 64 rows x 32 bf16 = 256 x 16B, 1 per thread
        copy16_g2l(&As[buf][ar][aq], &A[(size_t)(bm + ar) * K + k0 + aq]);
        // B tile: 256 cols x 32 bf16 = 1024 x 16B, 4 per thread
        #pragma unroll
        for (int t = 0; t < 4; t++) {
            const int idx = tid + t * 256;
            const int br = idx >> 2, bq = (idx & 3) * 8;
            copy16_g2l(&BsT[buf][br][bq], &Bt[(size_t)(bn + br) * K + k0 + bq]);
        }
    };

    stage(0, 0);
    int ibuf = 0;

    for (int k0 = 0; k0 < K; k0 += GTK) {
        copy16_wait();          // this wave's tile-k copies are in LDS
        __syncthreads();        // everyone's are; prior buffer is dead
        if (k0 + GTK < K) stage(k0 + GTK, ibuf ^ 1);   // overlap DMA with math

        // A fragments (ISA 16-bit A layout): lane<16 K{0-7,16-23}, lane>=16 K{8-15,24-31}
        v16bf afrag[2];
        #pragma unroll
        for (int t = 0; t < 2; t++) {
            const int row = wm * 32 + t * 16 + lh;
            const int kb  = hi ? 8 : 0;
            afrag[t] = load_frag(&As[ibuf][row][kb], &As[ibuf][row][kb + 16]);
        }
        // B fragments (matching A): lane<16 K{0-7,16-23}, lane>=16 K{8-15,24-31}, col=lane&15
        v16bf bfrag[4];
        #pragma unroll
        for (int t = 0; t < 4; t++) {
            const int col = wn * 64 + t * 16 + lh;
            const int kb  = hi ? 8 : 0;
            bfrag[t] = load_frag(&BsT[ibuf][col][kb], &BsT[ibuf][col][kb + 16]);
        }
        #pragma unroll
        for (int i = 0; i < 2; i++)
            #pragma unroll
            for (int j = 0; j < 4; j++)
                acc[i][j] = __builtin_amdgcn_wmma_f32_16x16x32_bf16(
                    false, afrag[i], false, bfrag[j], (short)0, acc[i][j],
                    false, false);
        ibuf ^= 1;
    }

    // --- Store: C/D layout; pointer-increment addressing ---
    const int rbase = hi ? 8 : 0;
    #pragma unroll
    for (int i = 0; i < 2; i++)
        #pragma unroll
        for (int j = 0; j < 4; j++) {
            float* p = C + (size_t)(bm + wm * 32 + i * 16 + rbase) * N
                         + (bn + wn * 64 + j * 16 + lh);
            #pragma unroll
            for (int e = 0; e < 8; e++) { *p = acc[i][j][e]; p += N; }
        }
}

// ---------------------------------------------------------------------------
// RoPE + scale + layout change: qkv f32 [B,S,H,3*HD] -> Qb/Kb/Vb bf16 [B,H,S,HD]
// ---------------------------------------------------------------------------
__global__ __launch_bounds__(256)
void rope_split_kernel(const float* __restrict__ qkv,
                       const int* __restrict__ segpos,
                       __bf16* __restrict__ Qb, __bf16* __restrict__ Kb,
                       __bf16* __restrict__ Vb)
{
    const int idx = blockIdx.x * 256 + threadIdx.x;   // 2^21 total
    const int i = idx & 31;
    const int h = (idx >> 5) & (HH - 1);
    const int s = (idx >> 9) & (SS - 1);
    const int b = idx >> 20;

    const float* row = qkv + ((size_t)(b * SS + s)) * (3 * DD) + (size_t)h * (3 * HD);

    const int pos = segpos[b * SS + s];
    const float inv_freq = __expf(-9.210340371976184f * (float)i * (1.0f / 32.0f));
    const float ang = (float)pos * inv_freq;
    float sn, cs;
    __sincosf(ang, &sn, &cs);

    const float q1 = row[i],       q2 = row[32 + i];
    const float k1 = row[64 + i],  k2 = row[96 + i];
    const float v1 = row[128 + i], v2 = row[160 + i];

    const size_t base = ((size_t)(b * HH + h) * SS + s) * HD;
    const float qs = 0.125f;  // 1/sqrt(64)
    Qb[base + i]      = (__bf16)((q1 * cs - q2 * sn) * qs);
    Qb[base + 32 + i] = (__bf16)((q1 * sn + q2 * cs) * qs);
    Kb[base + i]      = (__bf16)(k1 * cs - k2 * sn);
    Kb[base + 32 + i] = (__bf16)(k1 * sn + k2 * cs);
    Vb[base + i]      = (__bf16)v1;
    Vb[base + 32 + i] = (__bf16)v2;
}

// ---------------------------------------------------------------------------
// Causal flash attention. Grid: (S/128, H, B). 8 waves; wave w owns 16 q rows.
// Double-buffered K/V staging: K via async DMA, V transposed via VGPRs.
// ---------------------------------------------------------------------------
#define KVB 32
#define KROWPAD 8   // K row stride 72 elems = 144B (16B-aligned)
#define VROWPAD 8   // V^T row stride 40 elems = 80B (16B-aligned)

__global__ __launch_bounds__(256)
void flash_attn_kernel(const __bf16* __restrict__ Qb, const __bf16* __restrict__ Kb,
                       const __bf16* __restrict__ Vb, __bf16* __restrict__ attnb)
{
    __shared__ __bf16 Ks [2][KVB][HD + KROWPAD];    // [buf][key][hd]
    __shared__ __bf16 VsT[2][HD][KVB + VROWPAD];    // [buf][hd][key]
    __shared__ __bf16 Ps[8][16][32];                // per-wave P tile

    const int tid  = threadIdx.x;
    const int w    = tid >> 5;
    const int lane = tid & 31;
    const int lh   = lane & 15;
    const bool hi  = lane >= 16;

    const int b  = blockIdx.z;
    const int h  = blockIdx.y;
    const int q0 = blockIdx.x * 128;
    const int qw = q0 + w * 16;
    const size_t head_base = (size_t)(b * HH + h) * SS;

    // --- This wave's Q as two A fragments (HD chunks 0-31, 32-63) ---
    const __bf16* Qrow = Qb + (head_base + (qw + lh)) * HD;
    v16bf aq[2];
    #pragma unroll
    for (int kc = 0; kc < 2; kc++) {
        const int kb0 = kc * 32 + (hi ? 8 : 0);
        aq[kc] = load_frag(Qrow + kb0, Qrow + kb0 + 16);
    }

    float mrow[8], lrow[8];
    v8f o[4] = {};
    #pragma unroll
    for (int e = 0; e < 8; e++) { mrow[e] = -1e30f; lrow[e] = 0.0f; }

    const int nkb = (q0 + 128) / KVB;
    const int rbase = hi ? 8 : 0;

    // Per-thread staging coordinates
    const int skey = tid >> 3;
    const int sd0  = (tid & 7) * 8;

    auto stage = [&](int kb, int buf) {
        const size_t g = (head_base + (kb * KVB + skey)) * HD + sd0;
        copy16_g2l(&Ks[buf][skey][sd0], Kb + g);      // K: async DMA
        union { uint4 u; __bf16 hsv[8]; } vv;         // V: transpose via VGPRs
        vv.u = *reinterpret_cast<const uint4*>(Vb + g);
        #pragma unroll
        for (int j = 0; j < 8; j++) VsT[buf][sd0 + j][skey] = vv.hsv[j];
    };

    stage(0, 0);
    int ibuf = 0;

    for (int kb = 0; kb < nkb; kb++) {
        copy16_wait();
        __syncthreads();        // tile kb published; prior buffer dead
        if (kb + 1 < nkb) stage(kb + 1, ibuf ^ 1);

        if (kb * KVB <= qw + 15) {
            // --- S = Q K^T : two 16x16 tiles (keys 0-15, 16-31 of block) ---
            v8f s[2];
            #pragma unroll
            for (int nt = 0; nt < 2; nt++) {
                v8f t = {};
                #pragma unroll
                for (int kc = 0; kc < 2; kc++) {
                    const int key = nt * 16 + lh;
                    const int kb2 = kc * 32 + (hi ? 8 : 0);
                    const v16bf bk = load_frag(&Ks[ibuf][key][kb2],
                                               &Ks[ibuf][key][kb2 + 16]);
                    t = __builtin_amdgcn_wmma_f32_16x16x32_bf16(
                        false, aq[kc], false, bk, (short)0, t, false, false);
                }
                s[nt] = t;
            }

            // --- Causal mask + online softmax ---
            const int col0 = kb * KVB + lh;
            float alpha[8];
            #pragma unroll
            for (int e = 0; e < 8; e++) {
                const int qr = qw + rbase + e;
                if (col0 > qr)      s[0][e] = -1e30f;
                if (col0 + 16 > qr) s[1][e] = -1e30f;
                float rm = fmaxf(s[0][e], s[1][e]);
                #pragma unroll
                for (int msk = 1; msk < 16; msk <<= 1)
                    rm = fmaxf(rm, __shfl_xor(rm, msk, 16));
                const float mn = fmaxf(mrow[e], rm);
                alpha[e] = __expf(mrow[e] - mn);
                mrow[e]  = mn;
                s[0][e] = __expf(s[0][e] - mn);
                s[1][e] = __expf(s[1][e] - mn);
                float rs = s[0][e] + s[1][e];
                #pragma unroll
                for (int msk = 1; msk < 16; msk <<= 1)
                    rs += __shfl_xor(rs, msk, 16);
                lrow[e] = lrow[e] * alpha[e] + rs;
            }
            #pragma unroll
            for (int t = 0; t < 4; t++)
                #pragma unroll
                for (int e = 0; e < 8; e++) o[t][e] *= alpha[e];

            // --- P (C/D layout) -> A layout via per-wave LDS tile ---
            __bf16* Pw = &Ps[w][0][0];
            #pragma unroll
            for (int e = 0; e < 8; e++) {
                const int row = rbase + e;
                Pw[row * 32 + lh]      = (__bf16)s[0][e];
                Pw[row * 32 + 16 + lh] = (__bf16)s[1][e];
            }
            const int kb3 = hi ? 8 : 0;
            const v16bf pa = load_frag(Pw + lh * 32 + kb3, Pw + lh * 32 + kb3 + 16);

            // --- O += P @ V : 4 WMMAs over HD tiles ---
            #pragma unroll
            for (int t = 0; t < 4; t++) {
                const int n   = t * 16 + lh;
                const int kb2 = hi ? 8 : 0;
                const v16bf bv = load_frag(&VsT[ibuf][n][kb2],
                                           &VsT[ibuf][n][kb2 + 16]);
                o[t] = __builtin_amdgcn_wmma_f32_16x16x32_bf16(
                    false, pa, false, bv, (short)0, o[t], false, false);
            }
        }
        ibuf ^= 1;
    }

    // --- Normalize, convert to bf16, write: attnb[b*S + qrow][h*64 + hd] ---
    float inv[8];
    #pragma unroll
    for (int e = 0; e < 8; e++) inv[e] = 1.0f / lrow[e];
    #pragma unroll
    for (int t = 0; t < 4; t++) {
        __bf16* p = attnb + ((size_t)(b * SS + qw + rbase)) * DD + h * HD + t * 16 + lh;
        #pragma unroll
        for (int e = 0; e < 8; e++) { *p = (__bf16)(o[t][e] * inv[e]); p += DD; }
    }
}

// ---------------------------------------------------------------------------
// Host-side launch
// ---------------------------------------------------------------------------
extern "C" void kernel_launch(void* const* d_in, const int* in_sizes, int n_in,
                              void* d_out, int out_size, void* d_ws, size_t ws_size,
                              hipStream_t stream)
{
    const float* x      = (const float*)d_in[0];   // [B,S,D]
    const int*   segpos = (const int*)d_in[1];     // [B,S]
    // d_in[2] = mask (causal, computed analytically)
    const float* W_in   = (const float*)d_in[3];   // [D, 3D]
    const float* W_out  = (const float*)d_in[4];   // [D, D]
    float* out = (float*)d_out;                    // [B,S,D]

    char* ws = (char*)d_ws;
    size_t off = 0;
    auto carve = [&](size_t bytes) -> void* {
        void* p = ws + off;
        off = (off + bytes + 255) & ~(size_t)255;
        return p;
    };
    const size_t MROWS = (size_t)BB * SS;                                 // 4096
    __bf16* xb    = (__bf16*)carve(MROWS * DD * sizeof(__bf16));          // 8 MB
    __bf16* Wint  = (__bf16*)carve((size_t)3 * DD * DD * sizeof(__bf16)); // 6 MB  [3D][D]
    __bf16* Woutt = (__bf16*)carve((size_t)DD * DD * sizeof(__bf16));     // 2 MB  [D][D]
    float*  qkv   = (float*)carve(MROWS * (3 * DD) * sizeof(float));      // 48 MB
    __bf16* Qb    = (__bf16*)carve(MROWS * DD * sizeof(__bf16));          // 8 MB
    __bf16* Kb    = (__bf16*)carve(MROWS * DD * sizeof(__bf16));
    __bf16* Vb    = (__bf16*)carve(MROWS * DD * sizeof(__bf16));
    __bf16* attnb = (__bf16*)carve(MROWS * DD * sizeof(__bf16));          // 8 MB
    (void)ws_size; (void)in_sizes; (void)n_in; (void)out_size;

    // 0) One-time prep: bf16 activations; bf16-transposed weights
    cvt_bf16_kernel<<<(MROWS * DD / 4) / 256, 256, 0, stream>>>(x, xb);
    transpose_cvt_kernel<<<dim3(3 * DD / 32, DD / 32), 256, 0, stream>>>(
        W_in, Wint, DD, 3 * DD);
    transpose_cvt_kernel<<<dim3(DD / 32, DD / 32), 256, 0, stream>>>(
        W_out, Woutt, DD, DD);

    // 1) QKV projection: [4096,1024] @ [1024,3072]
    gemm_bf16wmma<<<dim3((3 * DD) / GTN, MROWS / GTM), 256, 0, stream>>>(
        xb, Wint, qkv, (int)MROWS, 3 * DD, DD);

    // 2) RoPE + scale + head-major bf16 split
    rope_split_kernel<<<(BB * SS * HH * 32) / 256, 256, 0, stream>>>(
        qkv, segpos, Qb, Kb, Vb);

    // 3) Causal flash attention (bf16 output)
    flash_attn_kernel<<<dim3(SS / 128, HH, BB), 256, 0, stream>>>(Qb, Kb, Vb, attnb);

    // 4) Output projection: [4096,1024] @ [1024,1024]
    gemm_bf16wmma<<<dim3(DD / GTN, MROWS / GTM), 256, 0, stream>>>(
        attnb, Woutt, out, (int)MROWS, DD, DD);
}
